// Regression_11424613007859
// MI455X (gfx1250) — matrix operand order repaired and checked by reading, not compiled, not run on hardware
//
#include <hip/hip_runtime.h>
#include <hip/hip_bf16.h>
#include <math.h>

#define NUM_LAGS 9
#define HID 256
#define BSZ 32768
#define RROWS (BSZ * NUM_LAGS)   // 294912 rows of (sample,lag)

typedef __attribute__((ext_vector_type(16))) __bf16 v16bf;
typedef __attribute__((ext_vector_type(8)))  float  v8f;

union Frag32B { uint4 u[2]; v16bf v; };

// Row-normalized mean-aggregation matrix reconstructed from the DGL graph
// (windowed combinations with the last comb dropped, plus (7,8) and (8,0)).
constexpr float AGG[9][9] = {
  {0.f,      0.f,      0.f,  0.f,   0.f,   0.f,      0.f,      0.f,      1.f},
  {1.f,      0.f,      0.f,  0.f,   0.f,   0.f,      0.f,      0.f,      0.f},
  {1.f/3.f,  2.f/3.f,  0.f,  0.f,   0.f,   0.f,      0.f,      0.f,      0.f},
  {0.2f,     0.4f,     0.4f, 0.f,   0.f,   0.f,      0.f,      0.f,      0.f},
  {0.f,      0.2f,     0.4f, 0.4f,  0.f,   0.f,      0.f,      0.f,      0.f},
  {0.f,      0.f,      0.2f, 0.4f,  0.4f,  0.f,      0.f,      0.f,      0.f},
  {0.f,      0.f,      0.f,  0.2f,  0.4f,  0.4f,     0.f,      0.f,      0.f},
  {0.f,      0.f,      0.f,  0.f,   0.25f, 0.5f,     0.25f,    0.f,      0.f},
  {0.f,      0.f,      0.f,  0.f,   0.f,   1.f/3.f,  1.f/3.f,  1.f/3.f,  0.f},
};

__device__ __forceinline__ __bf16 f2bf(float f) { return (__bf16)f; }
__device__ __forceinline__ float  bf2f(__bf16 b) { return (float)b; }

// ---------------- weight transpose: W[K,N] f32 -> Wt[N,K] bf16 ----------------
__global__ void transpose_to_bf16(const float* __restrict__ W,
                                  __bf16* __restrict__ Wt, int K, int N) {
  int idx = blockIdx.x * 256 + threadIdx.x;      // idx = n*K + k
  if (idx >= K * N) return;
  int n = idx / K;
  int k = idx - n * K;
  Wt[idx] = f2bf(W[(size_t)k * N + n]);
}

// ---------------- branch init: l0 and w0 in bf16 ----------------
__global__ __launch_bounds__(256)
void branch_init(const float* __restrict__ lags, const float* __restrict__ weather,
                 const float* __restrict__ Wl1, const float* __restrict__ bl1,
                 const float* __restrict__ gbn, const float* __restrict__ bbn,
                 const float* __restrict__ Ww1, const float* __restrict__ bw1,
                 __bf16* __restrict__ l0, __bf16* __restrict__ w0) {
  size_t idx = (size_t)blockIdx.x * 256 + threadIdx.x;  // over R*256
  int r = (int)(idx >> 8);
  int h = (int)(idx & 255);
  // lags branch: Linear(1,H) -> tanh -> eval-BN
  float x  = lags[r];
  float lv = tanhf(x * Wl1[h] + bl1[h]);
  lv = lv * (gbn[h] * rsqrtf(1.0f + 1e-5f)) + bbn[h];
  l0[idx] = f2bf(lv);
  // weather branch: Linear(8,H) -> tanh
  const float* wt = weather + (size_t)r * 8;
  float acc = bw1[h];
#pragma unroll
  for (int j = 0; j < 8; ++j) acc += wt[j] * Ww1[j * 256 + h];
  w0[idx] = f2bf(tanhf(acc));
}

// ---------------- WMMA GEMM: C[M,N] = act(A[M,K] @ Bt[N,K]^T + bias) ----------
// Block: 256 threads (8 waves). Block tile: 256 rows x 64 cols.
// Each wave: two 16-row M tiles x 64 cols -> 8 WMMAs per k-step reusing 4 B frags.
// B strip (64 x K) staged in LDS (padded rows, bank-conflict-free b128 reads).
// A fragments double-buffered from global; B fragments batch-loaded into
// distinct registers so ds waits drain incrementally instead of per-WMMA.
template <bool TANH, int K>
__global__ __launch_bounds__(256)
void gemm_bf16(const __bf16* __restrict__ A, const __bf16* __restrict__ Bt,
               const float* __restrict__ bias, __bf16* __restrict__ C,
               int M, int N) {
  constexpr int KP = K + 8;                 // padded LDS row stride (elements)
  __shared__ __bf16 sB[64 * KP];

  const int lane = threadIdx.x & 31;
  const int wv   = threadIdx.x >> 5;
  const int hi   = lane >> 4;   // half-wave select
  const int lm   = lane & 15;
  const int m0 = (blockIdx.y * 8 + wv) * 32;   // two 16-row tiles: m0, m0+16
  const int n0 = blockIdx.x * 64;

  // Cooperative fill of the 64 x K B strip (b128 granules).
  {
    constexpr int ELEMS = 64 * K;
#pragma unroll
    for (int i = threadIdx.x * 8; i < ELEMS; i += 256 * 8) {
      int n = i / K;
      int k = i - n * K;
      *(uint4*)&sB[n * KP + k] = *(const uint4*)&Bt[(size_t)(n0 + n) * K + k];
    }
  }
  __syncthreads();

  const __bf16* rowA0 = A + (size_t)(m0 + lm) * K;
  const __bf16* rowA1 = A + (size_t)(m0 + 16 + lm) * K;

  v8f acc[2][4];
#pragma unroll
  for (int mi = 0; mi < 2; ++mi)
#pragma unroll
    for (int t = 0; t < 4; ++t)
#pragma unroll
      for (int e = 0; e < 8; ++e) acc[mi][t][e] = 0.0f;

  // A fragment (ISA 16-bit A layout): lane<16 -> K {0..7,16..23}; lane>=16 -> {8..15,24..31}
  Frag32B a_cur[2];
  a_cur[0].u[0] = *(const uint4*)(rowA0 + hi * 8);
  a_cur[0].u[1] = *(const uint4*)(rowA0 + 16 + hi * 8);
  a_cur[1].u[0] = *(const uint4*)(rowA1 + hi * 8);
  a_cur[1].u[1] = *(const uint4*)(rowA1 + 16 + hi * 8);

  for (int k0 = 0; k0 < K; k0 += 32) {
    // Prefetch next A fragments (redundant reload on the last iteration).
    const int kn = (k0 + 32 < K) ? (k0 + 32) : k0;
    Frag32B a_nxt[2];
    a_nxt[0].u[0] = *(const uint4*)(rowA0 + kn + hi * 8);
    a_nxt[0].u[1] = *(const uint4*)(rowA0 + kn + 16 + hi * 8);
    a_nxt[1].u[0] = *(const uint4*)(rowA1 + kn + hi * 8);
    a_nxt[1].u[1] = *(const uint4*)(rowA1 + kn + 16 + hi * 8);

    // Batch-load all 4 B fragments into distinct registers first.
    Frag32B b[4];
#pragma unroll
    for (int t = 0; t < 4; ++t) {
      const __bf16* pB = &sB[(t * 16 + lm) * KP + k0 + hi * 8];
      b[t].u[0] = *(const uint4*)(pB);
      b[t].u[1] = *(const uint4*)(pB + 16);
    }

    // 8 WMMAs per k-step: 2 M tiles x 4 N tiles, reusing b[t].
#pragma unroll
    for (int t = 0; t < 4; ++t)
      acc[0][t] = __builtin_amdgcn_wmma_f32_16x16x32_bf16(
          false, a_cur[0].v, false, b[t].v, (short)0, acc[0][t], false, false);
#pragma unroll
    for (int t = 0; t < 4; ++t)
      acc[1][t] = __builtin_amdgcn_wmma_f32_16x16x32_bf16(
          false, a_cur[1].v, false, b[t].v, (short)0, acc[1][t], false, false);

    a_cur[0] = a_nxt[0];
    a_cur[1] = a_nxt[1];
  }

  // D layout: lane<16 -> rows +0..7 ; lane>=16 -> rows +8..15 ; col = lane%16
#pragma unroll
  for (int mi = 0; mi < 2; ++mi)
#pragma unroll
    for (int t = 0; t < 4; ++t) {
      int col = n0 + t * 16 + lm;
      float bv = bias[col];
#pragma unroll
      for (int e = 0; e < 8; ++e) {
        int row = m0 + mi * 16 + hi * 8 + e;
        float v = acc[mi][t][e] + bv;
        if (TANH) v = tanhf(v);
        C[(size_t)row * N + col] = f2bf(v);
      }
    }
}

// ---------------- mean-aggregate + concat: cat[b*9+d] = [A@l | A@w] ----------
__global__ __launch_bounds__(256)
void aggregate(const __bf16* __restrict__ L, int ldl,
               const __bf16* __restrict__ W, int ldw,
               __bf16* __restrict__ cat) {
  size_t idx = (size_t)blockIdx.x * 256 + threadIdx.x;  // over BSZ*HID
  int b = (int)(idx >> 8);
  int h = (int)(idx & 255);
  float lv[9], wv[9];
#pragma unroll
  for (int s = 0; s < 9; ++s) {
    lv[s] = bf2f(L[(size_t)(b * 9 + s) * ldl + h]);
    wv[s] = bf2f(W[(size_t)(b * 9 + s) * ldw + h]);
  }
#pragma unroll
  for (int d = 0; d < 9; ++d) {
    float ml = 0.f, mw = 0.f;
#pragma unroll
    for (int s = 0; s < 9; ++s) {   // AGG is constexpr -> zero terms fold away
      ml += AGG[d][s] * lv[s];
      mw += AGG[d][s] * wv[s];
    }
    size_t ro = (size_t)(b * 9 + d) * 512;
    cat[ro + h]       = f2bf(ml);
    cat[ro + 256 + h] = f2bf(mw);
  }
}

// ---------------- final regression: out[b] = l3[b,:] . W_reg + b_reg ---------
__global__ __launch_bounds__(256)
void reduce_reg(const __bf16* __restrict__ L3, const float* __restrict__ Wreg,
                const float* __restrict__ breg, float* __restrict__ out) {
  int wv = threadIdx.x >> 5, lane = threadIdx.x & 31;
  int b = blockIdx.x * 8 + wv;
  const __bf16* row = L3 + (size_t)b * (NUM_LAGS * HID);
  float acc = 0.f;
#pragma unroll 4
  for (int i = lane; i < NUM_LAGS * HID; i += 32) acc += bf2f(row[i]) * Wreg[i];
#pragma unroll
  for (int off = 16; off > 0; off >>= 1) acc += __shfl_xor(acc, off, 32);
  if (lane == 0) out[b] = acc + breg[0];
}

extern "C" void kernel_launch(void* const* d_in, const int* in_sizes, int n_in,
                              void* d_out, int out_size, void* d_ws, size_t ws_size,
                              hipStream_t stream) {
  const float* lags    = (const float*)d_in[0];
  const float* weather = (const float*)d_in[1];
  const float* W_l1    = (const float*)d_in[2];
  const float* b_l1    = (const float*)d_in[3];
  const float* g_bn    = (const float*)d_in[4];
  const float* b_bn    = (const float*)d_in[5];
  const float* W_l2    = (const float*)d_in[6];
  const float* b_l2    = (const float*)d_in[7];
  const float* W_w1    = (const float*)d_in[8];
  const float* b_w1    = (const float*)d_in[9];
  const float* W_w2    = (const float*)d_in[10];
  const float* b_w2    = (const float*)d_in[11];
  const float* Wm1     = (const float*)d_in[12];
  const float* bm1     = (const float*)d_in[13];
  const float* Wm2     = (const float*)d_in[14];
  const float* bm2     = (const float*)d_in[15];
  const float* W_reg   = (const float*)d_in[16];
  const float* b_reg   = (const float*)d_in[17];
  float* out = (float*)d_out;

  const size_t RH = (size_t)RROWS * HID;
  __bf16* base = (__bf16*)d_ws;
  __bf16* bufA = base;              // l0, later l2
  __bf16* bufB = bufA + RH;         // w0, later l3
  __bf16* bufC = bufB + RH;         // l1
  __bf16* bufD = bufC + RH;         // w1
  __bf16* cat1 = bufD + RH;         // [R,512] = [ml1|mw1]
  __bf16* cat2 = cat1 + 2 * RH;     // [R,512] = [ml2|mw2]
  __bf16* Wl2t = cat2 + 2 * RH;     // [256,256]
  __bf16* Ww2t = Wl2t + 256 * 256;
  __bf16* Wm1t = Ww2t + 256 * 256;  // [256,512]
  __bf16* Wm2t = Wm1t + 512 * 256;

  transpose_to_bf16<<<(256 * 256 + 255) / 256, 256, 0, stream>>>(W_l2, Wl2t, 256, 256);
  transpose_to_bf16<<<(256 * 256 + 255) / 256, 256, 0, stream>>>(W_w2, Ww2t, 256, 256);
  transpose_to_bf16<<<(512 * 256 + 255) / 256, 256, 0, stream>>>(Wm1, Wm1t, 512, 256);
  transpose_to_bf16<<<(512 * 256 + 255) / 256, 256, 0, stream>>>(Wm2, Wm2t, 512, 256);

  branch_init<<<RROWS, 256, 0, stream>>>(lags, weather, W_l1, b_l1, g_bn, b_bn,
                                         W_w1, b_w1, bufA, bufB);

  dim3 gg(HID / 64, RROWS / 256);
  gemm_bf16<true, 256><<<gg, 256, 0, stream>>>(bufA, Wl2t, b_l2, bufC, RROWS, HID);
  gemm_bf16<true, 256><<<gg, 256, 0, stream>>>(bufB, Ww2t, b_w2, bufD, RROWS, HID);

  aggregate<<<(BSZ * HID) / 256, 256, 0, stream>>>(bufC, HID, bufD, HID, cat1);

  gemm_bf16<false, 512><<<gg, 256, 0, stream>>>(cat1, Wm1t, bm1, bufA, RROWS, HID);

  aggregate<<<(BSZ * HID) / 256, 256, 0, stream>>>(bufA, HID, cat1 + 256, 512, cat2);

  gemm_bf16<false, 512><<<gg, 256, 0, stream>>>(cat2, Wm2t, bm2, bufB, RROWS, HID);

  reduce_reg<<<BSZ / 8, 256, 0, stream>>>(bufB, W_reg, b_reg, out);
}
